// PathScorerGNN_80796924772859
// MI455X (gfx1250) — matrix operand order repaired and checked by reading, not compiled, not run on hardware
//
#include <hip/hip_runtime.h>
#include <hip/hip_bf16.h>

// ---------------------------------------------------------------------------
// Problem constants (from reference): N nodes, E edges, B graphs, L path len,
// D input dim, H hidden dim.
// ---------------------------------------------------------------------------
#define NN 50000
#define EE 800000
#define BB 64
#define LL 32
#define DD 256
#define HH 256

// Use gfx1250 async direct-to-LDS staging in the GEMM (set 0 to fall back to
// synchronous global_load + ds_store staging).
#define USE_ASYNC_STAGE 1

typedef _Float16 v8h  __attribute__((ext_vector_type(8)));
typedef _Float16 v16h __attribute__((ext_vector_type(16)));
typedef float    v8f  __attribute__((ext_vector_type(8)));

static __device__ __forceinline__ v16h cat8h(v8h lo, v8h hi) {
    return __builtin_shufflevector(lo, hi, 0,1,2,3,4,5,6,7,8,9,10,11,12,13,14,15);
}

static __device__ __forceinline__ void atomic_add_f32(float* p, float v) {
    __hip_atomic_fetch_add(p, v, __ATOMIC_RELAXED, __HIP_MEMORY_SCOPE_AGENT);
}

static __device__ __forceinline__ void async_wait0() {
#if __has_builtin(__builtin_amdgcn_s_wait_asynccnt)
    __builtin_amdgcn_s_wait_asynccnt(0);
#else
    asm volatile("s_wait_asynccnt 0" ::: "memory");
#endif
}

// ---------------------------------------------------------------------------
// Workspace layout (bytes). All offsets 16B aligned.
// ---------------------------------------------------------------------------
#define OFF_AGG      ((size_t)0)                      // N*256 f32 (reused as node_emb)
#define OFF_DEG      (OFF_AGG   + (size_t)NN*256*4)   // N f32
#define OFF_XH       (OFF_DEG   + (size_t)NN*4)       // N*256 f16 (x in f16)
#define OFF_MH       (OFF_XH    + (size_t)NN*256*2)   // N*256 f16 (mean in f16)
#define OFF_HB       (OFF_MH    + (size_t)NN*256*2)   // N*256 f16 (layer1 out, f16)
#define OFF_WC1      (OFF_HB    + (size_t)NN*256*2)   // packed [512x256] f16
#define OFF_WC2      (OFF_WC1   + (size_t)512*256*2)
#define OFF_WIHH     (OFF_WC2   + (size_t)512*256*2)  // packed [512x1024] f16
#define OFF_BSUM     (OFF_WIHH  + (size_t)512*1024*2) // 1024 f32
#define OFF_GSUM     (OFF_BSUM  + (size_t)1024*4)     // B*256 f32
#define OFF_GCNT     (OFF_GSUM  + (size_t)BB*256*4)   // B f32 (contiguous after GSUM)
#define OFF_PSEQ     (OFF_GCNT  + (size_t)BB*4 + 192) // B*L*256 f16 (pad to 16B align)
#define OFF_PEMB     (OFF_PSEQ  + (size_t)BB*LL*256*2)// B*256 f32

// ---------------------------------------------------------------------------
// Small elementwise kernels
// ---------------------------------------------------------------------------
__global__ void f32_to_f16_kernel(const float* __restrict__ in,
                                  _Float16* __restrict__ out, int n) {
    int i = blockIdx.x * blockDim.x + threadIdx.x;
    if (i < n) out[i] = (_Float16)in[i];
}

__global__ void mean_f16_kernel(const float* __restrict__ agg,
                                const float* __restrict__ deg,
                                _Float16* __restrict__ out, int n) {
    int i = blockIdx.x * blockDim.x + threadIdx.x;
    if (i < n) {
        float d = deg[i >> 8];
        d = d > 1.0f ? d : 1.0f;
        out[i] = (_Float16)(agg[i] / d);
    }
}

__global__ void count_kernel(const int* __restrict__ idx, float* __restrict__ cnt, int n) {
    int i = blockIdx.x * blockDim.x + threadIdx.x;
    if (i < n) atomic_add_f32(&cnt[idx[i]], 1.0f);
}

__global__ void bsum_kernel(const float* __restrict__ a, const float* __restrict__ b,
                            float* __restrict__ o) {
    int i = blockIdx.x * blockDim.x + threadIdx.x;
    if (i < 4 * HH) o[i] = a[i] + b[i];
}

// Scatter-add neighbor features: one wave per edge, 8 dwords per lane.
__global__ __launch_bounds__(256)
void scatter_kernel(const float* __restrict__ xf, const _Float16* __restrict__ xh,
                    const int* __restrict__ src, const int* __restrict__ dst,
                    float* __restrict__ agg, int nedges) {
    int gw   = (blockIdx.x * blockDim.x + threadIdx.x) >> 5;
    int lane = threadIdx.x & 31;
    if (gw >= nedges) return;
    int s = src[gw], d = dst[gw];
    const float* xs = xf ? (xf + (size_t)s * 256) : nullptr;
    const _Float16* hs = xh ? (xh + (size_t)s * 256) : nullptr;
    float* ad = agg + (size_t)d * 256;
    #pragma unroll
    for (int k = 0; k < 8; ++k) {
        int c = lane + k * 32;
        float v = xs ? xs[c] : (float)hs[c];
        atomic_add_f32(&ad[c], v);
    }
}

// Mean-pool accumulate: node_emb -> gsum[batch]
__global__ void pool_kernel(const float* __restrict__ emb, const int* __restrict__ batch,
                            float* __restrict__ gsum, int n) {
    int i = blockIdx.x * blockDim.x + threadIdx.x;
    if (i < n) {
        int node = i >> 8, c = i & 255;
        atomic_add_f32(&gsum[batch[node] * 256 + c], emb[i]);
    }
}

// Gather path sequences and convert to f16 for the LSTM WMMA.
__global__ void gather_path_kernel(const float* __restrict__ emb, const int* __restrict__ paths,
                                   _Float16* __restrict__ out, int n) {
    int i = blockIdx.x * blockDim.x + threadIdx.x;
    if (i < n) {
        int pl = i >> 8, c = i & 255;
        out[i] = (_Float16)emb[(size_t)paths[pl] * 256 + c];
    }
}

// ---------------------------------------------------------------------------
// Weight packing into WMMA B-fragment layout:
//   B 32x16 f16 fragment: lane (0..31) holds column n = nt*16 + (lane&15),
//   elements e<8 map to K = 8*(lane>>4) + e, e>=8 to K = 16 + 8*(lane>>4) + (e-8). We store each lane's 16
//   halfs contiguously: packed[((kt*NT + nt)*32 + lane)*16 + e].
//   K axis is the concat [W0 (k<256) ; W1 (k>=256)].
//   transpose=0: W is [256 x ncols] row-major  (SAGE weights)
//   transpose=1: W is [ncols x 256] row-major  (LSTM w_ih/w_hh, used as W^T)
// ---------------------------------------------------------------------------
__global__ void pack_w_kernel(const float* __restrict__ W0, const float* __restrict__ W1,
                              _Float16* __restrict__ out, int ncols, int transpose) {
    int t = blockIdx.x * blockDim.x + threadIdx.x;
    int total = 16 * (ncols / 16) * 32;       // Ktot=512 -> 16 kt tiles
    if (t >= total) return;
    int lane = t & 31;
    int nt   = (t >> 5) % (ncols / 16);
    int kt   = (t >> 5) / (ncols / 16);
    int n    = nt * 16 + (lane & 15);
    int lh   = lane >> 4;
    _Float16* dp = out + (size_t)t * 16;
    #pragma unroll
    for (int e = 0; e < 16; ++e) {
        int k = kt * 32 + ((e < 8) ? (8 * lh + e) : (16 + 8 * lh + (e - 8)));
        const float* W = (k < 256) ? W0 : W1;
        int kk = k & 255;
        float v = transpose ? W[(size_t)n * 256 + kk] : W[(size_t)kk * ncols + n];
        dp[e] = (_Float16)v;
    }
}

// ---------------------------------------------------------------------------
// Fused dual-source GEMM:  OUT[r, 0:256] = act([A0|A1][r, 0:512] @ Wp + bias)
// 256 threads (8 waves). Block computes 64 rows x 256 cols.
// Wave w: row tile rt = w&3, column half (w>>2)*128 (8 N-tiles).
// A tile (64x512 f16) staged row-major in LDS via async direct-to-LDS copies,
// then consumed as aligned ds_load_b128 fragments.
// ---------------------------------------------------------------------------
__global__ __launch_bounds__(256)
void gemm_dual_kernel(const _Float16* __restrict__ A0, const _Float16* __restrict__ A1,
                      const _Float16* __restrict__ Wp, const float* __restrict__ bias,
                      float* __restrict__ outF, _Float16* __restrict__ outH,
                      int nrows, int relu) {
    extern __shared__ char dynsm[];
    _Float16* Atile = (_Float16*)dynsm;            // 64 * 512 halfs = 64KB
    const int tid = threadIdx.x;
    const int blockRow = blockIdx.x * 64;

    // Cooperative stage: 4096 chunks of 8 halfs (16B), 16 per thread.
    #pragma unroll
    for (int i = 0; i < 16; ++i) {
        int ci  = tid + 256 * i;
        int row = ci >> 6;                         // 64 chunks per row
        int off = (ci & 63) * 8;                   // half offset in [0,512)
        int gr  = blockRow + row; if (gr >= nrows) gr = nrows - 1;
        const _Float16* sp = (off < 256) ? (A0 + (size_t)gr * 256 + off)
                                         : (A1 + (size_t)gr * 256 + (off - 256));
#if USE_ASYNC_STAGE
        // gfx1250 async copy: global -> LDS without VGPR round trip (ASYNCcnt).
        // LDS address operand = wave-relative byte offset (low 32 bits of the
        // generic shared pointer).
        unsigned ldsoff = (unsigned)(size_t)(&Atile[row * 512 + off]);
        asm volatile("global_load_async_to_lds_b128 %0, %1, off"
                     :: "v"(ldsoff), "v"(sp) : "memory");
#else
        *(v8h*)&Atile[row * 512 + off] = *(const v8h*)sp;
#endif
    }
#if USE_ASYNC_STAGE
    async_wait0();
#endif
    __syncthreads();

    const int lane  = tid & 31;
    const int w     = tid >> 5;
    const int rt    = w & 3;
    const int nhalf = w >> 2;
    const int m     = lane & 15;
    const int lh    = lane >> 4;

    v8f z = {0.f,0.f,0.f,0.f,0.f,0.f,0.f,0.f};
    v8f acc[8];
    #pragma unroll
    for (int j = 0; j < 8; ++j) acc[j] = z;

    for (int kt = 0; kt < 16; ++kt) {
        int ab = (rt * 16 + m) * 512 + kt * 32 + 8 * lh;
        v8h alo = *(const v8h*)&Atile[ab];
        v8h ahi = *(const v8h*)&Atile[ab + 16];
        // prefetch next K-chunk of B (wraps within packed buffer)
        {
            int ktn = (kt + 1) & 15;
            __builtin_prefetch(Wp + ((size_t)((ktn * 16 + nhalf * 8) * 32 + lane)) * 16, 0, 1);
        }
        // Preload all 8 B fragments (one big load clause), then issue the 8
        // WMMAs so loads drain while the matrix pipe is busy.
        v8h blo[8], bhi[8];
        #pragma unroll
        for (int j = 0; j < 8; ++j) {
            int nt = nhalf * 8 + j;
            const _Float16* bp = Wp + ((size_t)((kt * 16 + nt) * 32 + lane)) * 16;
            blo[j] = *(const v8h*)bp;
            bhi[j] = *(const v8h*)(bp + 8);
        }
        v16h a = cat8h(alo, ahi);
        #pragma unroll
        for (int j = 0; j < 8; ++j) {
            acc[j] = __builtin_amdgcn_wmma_f32_16x16x32_f16(
                false, a, false, cat8h(blo[j], bhi[j]), (short)0, acc[j], false, false);
        }
    }

    // Epilogue: C/D layout -> lane (0..15): N=lane, M=i ; lane>=16: N=lane-16, M=i+8
    #pragma unroll
    for (int j = 0; j < 8; ++j) {
        int col = nhalf * 128 + j * 16 + m;
        float bv = bias ? bias[col] : 0.0f;
        #pragma unroll
        for (int i = 0; i < 8; ++i) {
            int row = blockRow + rt * 16 + i + 8 * lh;
            if (row < nrows) {
                float v = acc[j][i] + bv;
                if (relu) v = v > 0.f ? v : 0.f;
                if (outF) outF[(size_t)row * 256 + col] = v;
                if (outH) outH[(size_t)row * 256 + col] = (_Float16)v;
            }
        }
    }
}

// ---------------------------------------------------------------------------
// Persistent single-block LSTM: 64 sequences x 32 steps x H=256.
// gates(64x1024) = [xt | h](64x512) @ Wp(512x1024) via WMMA each step.
// h (f16), c (f32), gates (f16) live in LDS (229KB of the 320KB WGP LDS).
// ---------------------------------------------------------------------------
__global__ __launch_bounds__(256)
void lstm_kernel(const _Float16* __restrict__ seq,   // [64][32][256] f16
                 const _Float16* __restrict__ Wp,    // packed K=512, Nc=1024
                 const float* __restrict__ bsum,     // [1024] = b_ih + b_hh
                 float* __restrict__ pathemb) {      // [64][256] f32
    extern __shared__ char dynsm[];
    float*    Cst = (float*)dynsm;                   // 64*256 f32  (64KB)
    _Float16* Hst = (_Float16*)(dynsm + 64 * 256 * 4);        // 64*256 f16 (32KB)
    _Float16* Gt  = (_Float16*)(dynsm + 64 * 256 * 4 + 64 * 256 * 2); // 64*1024 f16 (128KB)

    const int tid  = threadIdx.x;
    const int lane = tid & 31;
    const int w    = tid >> 5;
    const int rt   = w & 3;
    const int half = w >> 2;
    const int m    = lane & 15;
    const int lh   = lane >> 4;
    const int rr   = tid >> 2;            // elementwise row
    const int cc0  = (tid & 3) * 64;      // elementwise col base

    for (int i = tid; i < 64 * 256; i += 256) { Hst[i] = (_Float16)0.f; Cst[i] = 0.f; }
    __syncthreads();

    const v8f z = {0.f,0.f,0.f,0.f,0.f,0.f,0.f,0.f};
    const int arow = rt * 16 + m;

    for (int t = 0; t < LL; ++t) {
        // ---- gate GEMM: each wave does 32 N-tiles x 16 K-chunks ----
        for (int jt = 0; jt < 32; ++jt) {
            int ntg = half * 32 + jt;
            v8f acc = z;
            #pragma unroll
            for (int kt = 0; kt < 16; ++kt) {
                v8h alo, ahi;
                if (kt < 8) {   // xt from global f16 sequence
                    const _Float16* ap = seq + ((size_t)(arow * 32 + t) * 256 + kt * 32 + 8 * lh);
                    alo = *(const v8h*)ap; ahi = *(const v8h*)(ap + 16);
                } else {        // h from LDS
                    const _Float16* ap = Hst + (arow * 256 + (kt - 8) * 32 + 8 * lh);
                    alo = *(const v8h*)ap; ahi = *(const v8h*)(ap + 16);
                }
                v16h a = cat8h(alo, ahi);
                const _Float16* bp = Wp + ((size_t)((kt * 64 + ntg) * 32 + lane)) * 16;
                v16h b = cat8h(*(const v8h*)bp, *(const v8h*)(bp + 8));
                acc = __builtin_amdgcn_wmma_f32_16x16x32_f16(
                    false, a, false, b, (short)0, acc, false, false);
            }
            int col = ntg * 16 + m;
            float bv = bsum[col];
            #pragma unroll
            for (int i = 0; i < 8; ++i) {
                int row = rt * 16 + i + 8 * lh;
                Gt[row * 1024 + col] = (_Float16)(acc[i] + bv);
            }
        }
        __syncthreads();

        // ---- elementwise cell update (gate order i,f,g,o) ----
        for (int j = 0; j < 64; ++j) {
            int col = cc0 + j;
            float ig = (float)Gt[rr * 1024 + col];
            float fg = (float)Gt[rr * 1024 + col + 256];
            float gg = (float)Gt[rr * 1024 + col + 512];
            float og = (float)Gt[rr * 1024 + col + 768];
            float si = 1.f / (1.f + __expf(-ig));
            float sf = 1.f / (1.f + __expf(-fg));
            float so = 1.f / (1.f + __expf(-og));
            float c  = sf * Cst[rr * 256 + col] + si * tanhf(gg);
            Cst[rr * 256 + col] = c;
            Hst[rr * 256 + col] = (_Float16)(so * tanhf(c));
        }
        __syncthreads();
    }

    for (int i = tid; i < 64 * 256; i += 256) pathemb[i] = (float)Hst[i];
}

// ---------------------------------------------------------------------------
// Final scorer: combined = [graph_mean | path_emb] (64x512)
// scores = relu(combined @ wm1 + bm1) @ wm2 + bm2   -> d_out[64]
// ---------------------------------------------------------------------------
__global__ __launch_bounds__(256)
void scorer_kernel(const float* __restrict__ gsum, const float* __restrict__ gcnt,
                   const float* __restrict__ pathemb,
                   const float* __restrict__ wm1, const float* __restrict__ bm1,
                   const float* __restrict__ wm2, const float* __restrict__ bm2,
                   float* __restrict__ out) {
    extern __shared__ char dynsm[];
    float* hid = (float*)dynsm;                  // 64*256 f32
    int j = threadIdx.x;                          // hidden unit
    for (int r = 0; r < BB; ++r) {
        float cnt = gcnt[r]; cnt = cnt > 1.f ? cnt : 1.f;
        float inv = 1.f / cnt;
        float acc = bm1[j];
        for (int k = 0; k < 256; ++k)
            acc += (gsum[r * 256 + k] * inv) * wm1[(size_t)k * 256 + j];
        for (int k = 0; k < 256; ++k)
            acc += pathemb[r * 256 + k] * wm1[(size_t)(256 + k) * 256 + j];
        hid[r * 256 + j] = acc > 0.f ? acc : 0.f;
    }
    __syncthreads();
    if (j < BB) {
        float s = bm2[0];
        for (int k = 0; k < 256; ++k) s += hid[j * 256 + k] * wm2[k];
        out[j] = s;
    }
}

// ---------------------------------------------------------------------------
// Host-side launch orchestration (graph-capture safe: only async stream ops)
// ---------------------------------------------------------------------------
extern "C" void kernel_launch(void* const* d_in, const int* in_sizes, int n_in,
                              void* d_out, int out_size, void* d_ws, size_t ws_size,
                              hipStream_t stream) {
    const float* x     = (const float*)d_in[0];
    const int*   ei    = (const int*)d_in[1];
    const int*   batch = (const int*)d_in[2];
    const int*   paths = (const int*)d_in[3];
    const float* w1l   = (const float*)d_in[4];
    const float* b1l   = (const float*)d_in[5];
    const float* w1r   = (const float*)d_in[6];
    const float* w2l   = (const float*)d_in[7];
    const float* b2l   = (const float*)d_in[8];
    const float* w2r   = (const float*)d_in[9];
    const float* wih   = (const float*)d_in[10];
    const float* whh   = (const float*)d_in[11];
    const float* bih   = (const float*)d_in[12];
    const float* bhh   = (const float*)d_in[13];
    const float* wm1   = (const float*)d_in[14];
    const float* bm1   = (const float*)d_in[15];
    const float* wm2   = (const float*)d_in[16];
    const float* bm2   = (const float*)d_in[17];

    char* ws = (char*)d_ws;
    float*     AGG   = (float*)(ws + OFF_AGG);     // also node_emb after layer 2
    float*     DEG   = (float*)(ws + OFF_DEG);
    _Float16*  XH    = (_Float16*)(ws + OFF_XH);
    _Float16*  MH    = (_Float16*)(ws + OFF_MH);
    _Float16*  HB    = (_Float16*)(ws + OFF_HB);
    _Float16*  WC1   = (_Float16*)(ws + OFF_WC1);
    _Float16*  WC2   = (_Float16*)(ws + OFF_WC2);
    _Float16*  WIHH  = (_Float16*)(ws + OFF_WIHH);
    float*     BSUM  = (float*)(ws + OFF_BSUM);
    float*     GSUM  = (float*)(ws + OFF_GSUM);
    float*     GCNT  = (float*)(ws + OFF_GCNT);
    _Float16*  PSEQ  = (_Float16*)(ws + OFF_PSEQ);
    float*     PEMB  = (float*)(ws + OFF_PEMB);

    const int* src  = ei;
    const int* dstv = ei + EE;

    // zero accumulation buffers (every call; harness does not re-poison)
    hipMemsetAsync(AGG, 0, (size_t)NN * 256 * 4, stream);
    hipMemsetAsync(DEG, 0, (size_t)NN * 4, stream);
    hipMemsetAsync(GSUM, 0, ((size_t)BB * 256 + BB) * 4, stream);

    int n;
    // x -> f16
    n = NN * 256;
    f32_to_f16_kernel<<<(n + 255) / 256, 256, 0, stream>>>(x, XH, n);
    // pack weights into WMMA fragment layout
    n = 16 * 16 * 32;
    pack_w_kernel<<<(n + 255) / 256, 256, 0, stream>>>(w1l, w1r, WC1, 256, 0);
    pack_w_kernel<<<(n + 255) / 256, 256, 0, stream>>>(w2l, w2r, WC2, 256, 0);
    n = 16 * 64 * 32;
    pack_w_kernel<<<(n + 255) / 256, 256, 0, stream>>>(wih, whh, WIHH, 1024, 1);
    bsum_kernel<<<4, 256, 0, stream>>>(bih, bhh, BSUM);
    // degrees (edge) and graph counts (node)
    count_kernel<<<(EE + 255) / 256, 256, 0, stream>>>(dstv, DEG, EE);
    count_kernel<<<(NN + 255) / 256, 256, 0, stream>>>(batch, GCNT, NN);

    // ---- SAGE layer 1 ----
    scatter_kernel<<<(EE * 32 + 255) / 256, 256, 0, stream>>>(x, nullptr, src, dstv, AGG, EE);
    n = NN * 256;
    mean_f16_kernel<<<(n + 255) / 256, 256, 0, stream>>>(AGG, DEG, MH, n);
    gemm_dual_kernel<<<(NN + 63) / 64, 256, 64 * 512 * 2, stream>>>(
        MH, XH, WC1, b1l, nullptr, HB, NN, 1);

    // ---- SAGE layer 2 ----
    hipMemsetAsync(AGG, 0, (size_t)NN * 256 * 4, stream);
    scatter_kernel<<<(EE * 32 + 255) / 256, 256, 0, stream>>>(nullptr, HB, src, dstv, AGG, EE);
    mean_f16_kernel<<<(n + 255) / 256, 256, 0, stream>>>(AGG, DEG, MH, n);
    // node_emb written into AGG region (its old contents are no longer needed)
    gemm_dual_kernel<<<(NN + 63) / 64, 256, 64 * 512 * 2, stream>>>(
        MH, HB, WC2, b2l, AGG, nullptr, NN, 0);

    // ---- pooling + path gather ----
    pool_kernel<<<(NN * 256 + 255) / 256, 256, 0, stream>>>(AGG, batch, GSUM, NN * 256);
    n = BB * LL * 256;
    gather_path_kernel<<<(n + 255) / 256, 256, 0, stream>>>(AGG, paths, PSEQ, n);

    // ---- LSTM (single persistent block, 229KB LDS) ----
    lstm_kernel<<<1, 256, 64 * 256 * 4 + 64 * 256 * 2 + 64 * 1024 * 2, stream>>>(
        PSEQ, WIHH, BSUM, PEMB);

    // ---- scorer MLP ----
    scorer_kernel<<<1, 256, 64 * 256 * 4, stream>>>(
        GSUM, GCNT, PEMB, wm1, bm1, wm2, bm2, (float*)d_out);
}
